// WaveNetNN_16080357556388
// MI455X (gfx1250) — compile-verified, matrix operand order checked by reading, not run on hardware
//
#include <hip/hip_runtime.h>

// ---------------------------------------------------------------------------
// WaveNet forward, CDNA5 (gfx1250), wave32, bf16 WMMA 16x16x32 w/ f32 accum.
// Activations: bf16, [n][8192 (stride)][32ch], valid tau = 8196 + t.
// Layer i (d = 1<<(i%10)): out[t] depends on in[t], in[t+d]; only last 4096
// outputs needed => no padding semantics required (end-aligned equivalence).
// ---------------------------------------------------------------------------

typedef __bf16 bf16_t;
typedef __attribute__((ext_vector_type(8)))  float          v8f;
typedef __attribute__((ext_vector_type(16))) __bf16         v16bf;
typedef __attribute__((ext_vector_type(8)))  unsigned short ushort8;

union BF16x16 { v16bf v; bf16_t b[16]; ushort8 h[2]; };

// ---- workspace layout (bytes) ----
constexpr size_t SZ_X      = (size_t)8 * 8192 * 32 * 2;     // 4 MiB per buffer
constexpr size_t OFF_XA    = 0;
constexpr size_t OFF_XB    = OFF_XA + SZ_X;
constexpr size_t OFF_WSTART= OFF_XB + SZ_X;                  // 32x256 bf16
constexpr size_t OFF_WF    = OFF_WSTART + (size_t)32*256*2;  // 40x32x64 bf16
constexpr size_t OFF_WG    = OFF_WF + (size_t)40*32*64*2;
constexpr size_t OFF_WR    = OFF_WG + (size_t)40*32*64*2;    // 40x32x32 bf16
constexpr size_t OFF_WSKIP = OFF_WR + (size_t)40*32*32*2;    // 256x32 bf16
constexpr size_t OFF_W1    = OFF_WSKIP + (size_t)256*32*2;   // 256x256 bf16
constexpr size_t OFF_W2    = OFF_W1 + (size_t)256*256*2;

__device__ __forceinline__ float fsig(float x) {
    float e = __builtin_amdgcn_exp2f(-1.4426950408889634f * x);
    return __builtin_amdgcn_rcpf(1.0f + e);
}
__device__ __forceinline__ float ftanh(float x) {
    float e = __builtin_amdgcn_exp2f(2.8853900817779268f * x);   // e^{2x}
    return 1.0f - 2.0f * __builtin_amdgcn_rcpf(e + 1.0f);
}

// A/B fragment: caller passes pointer to (row_base + k0 + half*8); elements
// 0..7 <- [+0..+7], 8..15 <- [+16..+23]  (16-bit WMMA operand layout).
__device__ __forceinline__ v16bf frag_from_row(const bf16_t* p) {
    const ushort8* q = (const ushort8*)p;
    BF16x16 f;
    f.h[0] = q[0];
    f.h[1] = q[2];          // +16 elements
    return f.v;
}
// Strided fp32 gather (start conv: A[row=tau][k=class], k-stride = 16384)
__device__ __forceinline__ v16bf frag_gather_f32(const float* base, size_t kstride, int k0h) {
    BF16x16 f;
#pragma unroll
    for (int j = 0; j < 8; ++j) {
        f.b[j]     = (bf16_t)(base[(size_t)(k0h + j)      * kstride]);
        f.b[j + 8] = (bf16_t)(base[(size_t)(k0h + 16 + j) * kstride]);
    }
    return f.v;
}
__device__ __forceinline__ v8f wmma_bf16(v16bf a, v16bf b, v8f c) {
    return __builtin_amdgcn_wmma_f32_16x16x32_bf16(false, a, false, b, (short)0, c, false, false);
}

// ---------------------------------------------------------------------------
// Weight packing fp32 -> bf16 (fragment-friendly [out][k] row-major)
// ---------------------------------------------------------------------------
__global__ void pack_weights(const float* start_w, const float* filter_w,
                             const float* gate_w, const float* residual_w,
                             const float* skip_w, const float* end1_w,
                             const float* end2_w,
                             bf16_t* Wstart, bf16_t* Wf, bf16_t* Wg,
                             bf16_t* Wr, bf16_t* Wskip, bf16_t* W1,
                             bf16_t* W2) {
    const int NT = 352256;
    for (int idx = blockIdx.x * blockDim.x + threadIdx.x; idx < NT;
         idx += gridDim.x * blockDim.x) {
        if (idx < 8192) {                       // start_w [32][256]
            Wstart[idx] = (bf16_t)start_w[idx];
        } else if (idx < 90112) {               // filter: [io][kk] kk<32:tap0
            int r = idx - 8192; int io = r >> 6; int kk = r & 63;
            Wf[r] = (bf16_t)filter_w[(io * 32 + (kk & 31)) * 2 + (kk >> 5)];
        } else if (idx < 172032) {
            int r = idx - 90112; int io = r >> 6; int kk = r & 63;
            Wg[r] = (bf16_t)gate_w[(io * 32 + (kk & 31)) * 2 + (kk >> 5)];
        } else if (idx < 212992) {              // residual [io*32+c]
            int r = idx - 172032; Wr[r] = (bf16_t)residual_w[r];
        } else if (idx < 221184) {              // skip layer 39 only [256][32]
            int r = idx - 212992; Wskip[r] = (bf16_t)skip_w[39 * 256 * 32 + r];
        } else if (idx < 286720) {
            int r = idx - 221184; W1[r] = (bf16_t)end1_w[r];
        } else {
            int r = idx - 286720; W2[r] = (bf16_t)end2_w[r];
        }
    }
}

// ---------------------------------------------------------------------------
// Start conv: x0[n][t][o] = sum_c start_w[o][c] * in[n][c][8196+t], t<8188
// M = time rows, K = 256 classes, N = 32 channels. 16 WMMA per wave-tile.
// ---------------------------------------------------------------------------
__global__ void start_conv(const float* __restrict__ xin,
                           const bf16_t* __restrict__ Wstart,
                           bf16_t* __restrict__ xout) {
    const int wave = threadIdx.x >> 5, lane = threadIdx.x & 31;
    const int lh = (lane >> 4) & 1, lr = lane & 15;
    const int n = blockIdx.y;
    const int t0 = (blockIdx.x * 8 + wave) * 16;
    int tr = t0 + lr; if (tr > 8187) tr = 8187;
    const float* abase = xin + (size_t)n * 256 * 16384 + (8196 + tr);

    v8f acc0 = {}, acc1 = {};
#pragma unroll
    for (int k0 = 0; k0 < 256; k0 += 32) {
        v16bf A  = frag_gather_f32(abase, 16384, k0 + lh * 8);
        v16bf B0 = frag_from_row(Wstart + (size_t)lr * 256 + k0 + lh * 8);
        v16bf B1 = frag_from_row(Wstart + (size_t)(16 + lr) * 256 + k0 + lh * 8);
        acc0 = wmma_bf16(A, B0, acc0);
        acc1 = wmma_bf16(A, B1, acc1);
    }
#pragma unroll
    for (int v = 0; v < 8; ++v) {
        int t = t0 + v + 8 * lh;
        if (t < 8188) {
            size_t idx = ((size_t)n * 8192 + t) * 32;
            xout[idx + lr]      = (bf16_t)acc0[v];
            xout[idx + 16 + lr] = (bf16_t)acc1[v];
        }
    }
}

// ---------------------------------------------------------------------------
// One WaveNet layer: h = tanh(Wf.[x_t;x_{t+d}]) * sig(Wg.[...]);
// out = Wr.h + x_{t+d}  (or out = h on the last layer, for the skip path).
// Each wave owns 32 rows (2 sub-tiles) so all B fragments are loaded once
// and reused: 20 WMMA per 10 B-fragment loads.
// ---------------------------------------------------------------------------
__global__ void layer_kernel(const bf16_t* __restrict__ xin,
                             bf16_t* __restrict__ xout,
                             const bf16_t* __restrict__ Wf_i,
                             const bf16_t* __restrict__ Wg_i,
                             const bf16_t* __restrict__ Wr_i,
                             int d, int Lout, int store_h) {
    __shared__ __align__(16) bf16_t hls[8][16 * 40];
    const int wave = threadIdx.x >> 5, lane = threadIdx.x & 31;
    const int lh = (lane >> 4) & 1, lr = lane & 15;
    const int n = blockIdx.y;
    const int t00 = (blockIdx.x * 8 + wave) * 32;
    const bf16_t* xn = xin + (size_t)n * 8192 * 32;

    // weight fragments: loaded once, reused by both sub-tiles
    v16bf Bf00 = frag_from_row(Wf_i + (size_t)lr * 64 + lh * 8);
    v16bf Bf01 = frag_from_row(Wf_i + (size_t)lr * 64 + 32 + lh * 8);
    v16bf Bf10 = frag_from_row(Wf_i + (size_t)(16 + lr) * 64 + lh * 8);
    v16bf Bf11 = frag_from_row(Wf_i + (size_t)(16 + lr) * 64 + 32 + lh * 8);
    v16bf Bg00 = frag_from_row(Wg_i + (size_t)lr * 64 + lh * 8);
    v16bf Bg01 = frag_from_row(Wg_i + (size_t)lr * 64 + 32 + lh * 8);
    v16bf Bg10 = frag_from_row(Wg_i + (size_t)(16 + lr) * 64 + lh * 8);
    v16bf Bg11 = frag_from_row(Wg_i + (size_t)(16 + lr) * 64 + 32 + lh * 8);
    v16bf Br0  = frag_from_row(Wr_i + (size_t)lr * 32 + lh * 8);
    v16bf Br1  = frag_from_row(Wr_i + (size_t)(16 + lr) * 32 + lh * 8);
    bf16_t* hw = &hls[wave][0];

#pragma unroll
    for (int s = 0; s < 2; ++s) {
        const int t0 = t00 + s * 16;
        int r1 = t0 + lr;      if (r1 > 8191) r1 = 8191;
        int r2 = t0 + lr + d;  if (r2 > 8191) r2 = 8191;

        v16bf A0 = frag_from_row(xn + (size_t)r1 * 32 + lh * 8);   // tap t
        v16bf A1 = frag_from_row(xn + (size_t)r2 * 32 + lh * 8);   // tap t+d

        v8f cf0 = {}, cf1 = {}, cg0 = {}, cg1 = {};
        cf0 = wmma_bf16(A0, Bf00, cf0); cf0 = wmma_bf16(A1, Bf01, cf0);
        cf1 = wmma_bf16(A0, Bf10, cf1); cf1 = wmma_bf16(A1, Bf11, cf1);
        cg0 = wmma_bf16(A0, Bg00, cg0); cg0 = wmma_bf16(A1, Bg01, cg0);
        cg1 = wmma_bf16(A0, Bg10, cg1); cg1 = wmma_bf16(A1, Bg11, cg1);

        float h0[8], h1[8];
#pragma unroll
        for (int v = 0; v < 8; ++v) {
            h0[v] = ftanh(cf0[v]) * fsig(cg0[v]);
            h1[v] = ftanh(cf1[v]) * fsig(cg1[v]);
        }

        if (store_h) {  // last layer: keep gated output for skip path
#pragma unroll
            for (int v = 0; v < 8; ++v) {
                int t = t0 + v + 8 * lh;
                if (t < Lout) {
                    size_t idx = ((size_t)n * 8192 + t) * 32;
                    xout[idx + lr]      = (bf16_t)h0[v];
                    xout[idx + 16 + lr] = (bf16_t)h1[v];
                }
            }
        } else {
            // C-layout -> A-layout relayout via per-wave LDS tile [16][40]
#pragma unroll
            for (int v = 0; v < 8; ++v) {
                int row = v + 8 * lh;
                hw[row * 40 + lr]      = (bf16_t)h0[v];
                hw[row * 40 + 16 + lr] = (bf16_t)h1[v];
            }
            __syncthreads();

            v16bf HA = frag_from_row(hw + (size_t)lr * 40 + lh * 8);

            v8f c0, c1;   // residual accumulators init'd with x_{t+d}
#pragma unroll
            for (int v = 0; v < 8; ++v) {
                int rr = t0 + v + 8 * lh + d; if (rr > 8191) rr = 8191;
                c0[v] = (float)xn[(size_t)rr * 32 + lr];
                c1[v] = (float)xn[(size_t)rr * 32 + 16 + lr];
            }
            c0 = wmma_bf16(HA, Br0, c0);
            c1 = wmma_bf16(HA, Br1, c1);

#pragma unroll
            for (int v = 0; v < 8; ++v) {
                int t = t0 + v + 8 * lh;
                if (t < Lout) {
                    size_t idx = ((size_t)n * 8192 + t) * 32;
                    xout[idx + lr]      = (bf16_t)c0[v];
                    xout[idx + 16 + lr] = (bf16_t)c1[v];
                }
            }
        }
    }
}

// ---------------------------------------------------------------------------
// Fused tail: out = end2.relu(end1.relu(skip.h) + b1) + b2 ; 16 rows / block,
// 8 waves x 2 N-tiles each; LDS staging between GEMM stages.
// ---------------------------------------------------------------------------
__global__ void end_kernel(const bf16_t* __restrict__ h,
                           const bf16_t* __restrict__ Wskip,
                           const bf16_t* __restrict__ W1,
                           const bf16_t* __restrict__ W2,
                           const float* __restrict__ b1,
                           const float* __restrict__ b2,
                           float* __restrict__ out) {
    __shared__ __align__(16) bf16_t Y[16 * 264];
    __shared__ __align__(16) bf16_t Y2[16 * 264];
    const int wave = threadIdx.x >> 5, lane = threadIdx.x & 31;
    const int lh = (lane >> 4) & 1, lr = lane & 15;
    const int m0 = blockIdx.x * 16;
    const int n = m0 >> 12, p0 = m0 & 4095;
    const int n0a = wave * 32, n0b = wave * 32 + 16;
    const bf16_t* hb = h + ((size_t)n * 8192 + p0) * 32;

    // stage 1: skip conv (K=32) + relu
    v16bf A = frag_from_row(hb + (size_t)lr * 32 + lh * 8);
    v16bf S0 = frag_from_row(Wskip + (size_t)(n0a + lr) * 32 + lh * 8);
    v16bf S1 = frag_from_row(Wskip + (size_t)(n0b + lr) * 32 + lh * 8);
    v8f s0 = {}, s1 = {};
    s0 = wmma_bf16(A, S0, s0);
    s1 = wmma_bf16(A, S1, s1);
#pragma unroll
    for (int v = 0; v < 8; ++v) {
        int row = v + 8 * lh;
        Y[row * 264 + n0a + lr] = (bf16_t)fmaxf(s0[v], 0.0f);
        Y[row * 264 + n0b + lr] = (bf16_t)fmaxf(s1[v], 0.0f);
    }
    __syncthreads();

    // stage 2: end1 (K=256) + bias + relu; prefetch end2 weights meanwhile
    __builtin_prefetch(W2 + (size_t)(n0a + lr) * 256, 0, 1);
    __builtin_prefetch(W2 + (size_t)(n0b + lr) * 256, 0, 1);
    v8f e0 = {}, e1 = {};
#pragma unroll
    for (int k0 = 0; k0 < 256; k0 += 32) {
        v16bf Af = frag_from_row(&Y[(size_t)lr * 264 + k0 + lh * 8]);
        v16bf B0 = frag_from_row(W1 + (size_t)(n0a + lr) * 256 + k0 + lh * 8);
        v16bf B1 = frag_from_row(W1 + (size_t)(n0b + lr) * 256 + k0 + lh * 8);
        e0 = wmma_bf16(Af, B0, e0);
        e1 = wmma_bf16(Af, B1, e1);
    }
    float bv0 = b1[n0a + lr], bv1 = b1[n0b + lr];
#pragma unroll
    for (int v = 0; v < 8; ++v) {
        int row = v + 8 * lh;
        Y2[row * 264 + n0a + lr] = (bf16_t)fmaxf(e0[v] + bv0, 0.0f);
        Y2[row * 264 + n0b + lr] = (bf16_t)fmaxf(e1[v] + bv1, 0.0f);
    }
    __syncthreads();

    // stage 3: end2 (K=256) + bias -> fp32 output [m][class]
    v8f f0 = {}, f1 = {};
#pragma unroll
    for (int k0 = 0; k0 < 256; k0 += 32) {
        v16bf Af = frag_from_row(&Y2[(size_t)lr * 264 + k0 + lh * 8]);
        v16bf B0 = frag_from_row(W2 + (size_t)(n0a + lr) * 256 + k0 + lh * 8);
        v16bf B1 = frag_from_row(W2 + (size_t)(n0b + lr) * 256 + k0 + lh * 8);
        f0 = wmma_bf16(Af, B0, f0);
        f1 = wmma_bf16(Af, B1, f1);
    }
    float cv0 = b2[n0a + lr], cv1 = b2[n0b + lr];
#pragma unroll
    for (int v = 0; v < 8; ++v) {
        int m = m0 + v + 8 * lh;
        out[(size_t)m * 256 + n0a + lr] = f0[v] + cv0;
        out[(size_t)m * 256 + n0b + lr] = f1[v] + cv1;
    }
}

// ---------------------------------------------------------------------------
extern "C" void kernel_launch(void* const* d_in, const int* in_sizes, int n_in,
                              void* d_out, int out_size, void* d_ws, size_t ws_size,
                              hipStream_t stream) {
    const float* x_input    = (const float*)d_in[0];
    const float* start_w    = (const float*)d_in[1];
    const float* filter_w   = (const float*)d_in[2];
    const float* gate_w     = (const float*)d_in[3];
    const float* residual_w = (const float*)d_in[4];
    const float* skip_w     = (const float*)d_in[5];
    const float* end1_w     = (const float*)d_in[6];
    const float* end1_b     = (const float*)d_in[7];
    const float* end2_w     = (const float*)d_in[8];
    const float* end2_b     = (const float*)d_in[9];

    char* ws = (char*)d_ws;
    bf16_t* xA     = (bf16_t*)(ws + OFF_XA);
    bf16_t* xB     = (bf16_t*)(ws + OFF_XB);
    bf16_t* Wstart = (bf16_t*)(ws + OFF_WSTART);
    bf16_t* Wf     = (bf16_t*)(ws + OFF_WF);
    bf16_t* Wg     = (bf16_t*)(ws + OFF_WG);
    bf16_t* Wr     = (bf16_t*)(ws + OFF_WR);
    bf16_t* Wskip  = (bf16_t*)(ws + OFF_WSKIP);
    bf16_t* W1     = (bf16_t*)(ws + OFF_W1);
    bf16_t* W2     = (bf16_t*)(ws + OFF_W2);

    pack_weights<<<dim3(344), dim3(256), 0, stream>>>(
        start_w, filter_w, gate_w, residual_w, skip_w, end1_w, end2_w,
        Wstart, Wf, Wg, Wr, Wskip, W1, W2);

    start_conv<<<dim3(64, 8), dim3(256), 0, stream>>>(x_input, Wstart, xA);

    int S = 0;
    for (int i = 0; i < 40; ++i) {
        int d = 1 << (i % 10);
        int Lout = 8188 - (S + d);
        const bf16_t* in = (i & 1) ? xB : xA;
        bf16_t* outb     = (i & 1) ? xA : xB;
        int nt = (Lout + 31) / 32;           // 32 rows per wave
        int gx = (nt + 7) / 8;
        layer_kernel<<<dim3(gx, 8), dim3(256), 0, stream>>>(
            in, outb, Wf + (size_t)i * 2048, Wg + (size_t)i * 2048,
            Wr + (size_t)i * 1024, d, Lout, (i == 39) ? 1 : 0);
        S += d;
    }
    // layer 39 (odd) wrote h into xA; skip/end pipeline on last 4096 steps.
    end_kernel<<<dim3(2048), dim3(256), 0, stream>>>(
        xA, Wskip, W1, W2, end1_b, end2_b, (float*)d_out);
}
